// MultiHeadGeometryAttention_89464168776146
// MI455X (gfx1250) — matrix operand order repaired and checked by reading, not compiled, not run on hardware
//
#include <hip/hip_runtime.h>

typedef _Float16 f16;
typedef __attribute__((ext_vector_type(16))) _Float16 v16h;
typedef __attribute__((ext_vector_type(8)))  float    v8f;
typedef __attribute__((ext_vector_type(4)))  float    f32x4;
typedef __attribute__((ext_vector_type(2)))  float    f32x2;
typedef __attribute__((ext_vector_type(4)))  unsigned u32x4;

#define BB 2
#define NN 2048
#define DD 128
#define HH 4
#define HD 32
#define JSPLIT 2
#define JCHUNK (NN / JSPLIT)   // 1024

// POD unions for WMMA operand staging (ext_vector types only)
union U16h { v16h v; u32x4 q[2]; unsigned u[8]; f16 h[16]; };
union Pack8 { f16 h[8]; u32x4 q; };
union Pack2 { f16 h[2]; unsigned u; };
union PackF8 { float f[8]; f32x4 q[2]; };

static __device__ __forceinline__ v8f wmma16(v16h a, v16h b, v8f c) {
  // v_wmma_f32_16x16x32_f16: D = A(16x32) x B(32x16) + C
  return __builtin_amdgcn_wmma_f32_16x16x32_f16(false, a, false, b, (short)0, c, false, false);
}

// ---------------------------------------------------------------------------
// Kernel 0: prep — transpose+convert weights to f16, pack per-point geometry.
// geomI[bn] = {x,y,z, px,py,pz, nx,ny,nz, sx,sy,sz, curv, dens, lin, 0} (16 f32)
// geomJ[bn] = {x,y,z, nx,ny,nz, curv, dens}                              (8 f32)
// ---------------------------------------------------------------------------
__global__ void prep_kernel(const float* __restrict__ x, const float* __restrict__ pdir,
                            const float* __restrict__ curv, const float* __restrict__ dens,
                            const float* __restrict__ nrm,  const float* __restrict__ lin,
                            const float* __restrict__ qkvw, const float* __restrict__ outw,
                            f16* __restrict__ wqkvT, f16* __restrict__ owT,
                            float* __restrict__ geomI, float* __restrict__ geomJ)
{
  const int RQ = 384 * 128;          // 49152
  const int RO = 128 * 128;          // 16384
  const int RG = BB * NN;            // 4096
  int idx = blockIdx.x * blockDim.x + threadIdx.x;
  if (idx < RQ) {
    int n = idx >> 7, k = idx & 127;
    wqkvT[n * 128 + k] = (f16)qkvw[k * 384 + n];
  } else if (idx < RQ + RO) {
    int t = idx - RQ;
    int n = t >> 7, k = t & 127;
    owT[n * 128 + k] = (f16)outw[k * 128 + n];
  } else if (idx < RQ + RO + RG) {
    int bn = idx - (RQ + RO);
    float px = pdir[bn*3+0], py = pdir[bn*3+1], pz = pdir[bn*3+2];
    float nx = nrm[bn*3+0],  ny = nrm[bn*3+1],  nz = nrm[bn*3+2];
    float sx = py*nz - pz*ny, sy = pz*nx - px*nz, sz = px*ny - py*nx;
    float inv = 1.0f / (sqrtf(sx*sx + sy*sy + sz*sz) + 1e-8f);
    sx *= inv; sy *= inv; sz *= inv;
    float* g = geomI + (size_t)bn * 16;
    g[0] = x[(size_t)bn*DD + 0]; g[1] = x[(size_t)bn*DD + 1]; g[2] = x[(size_t)bn*DD + 2];
    g[3] = px; g[4] = py; g[5] = pz;
    g[6] = nx; g[7] = ny; g[8] = nz;
    g[9] = sx; g[10] = sy; g[11] = sz;
    g[12] = curv[bn]; g[13] = dens[bn]; g[14] = lin[bn]; g[15] = 0.0f;
  } else if (idx < RQ + RO + 2 * RG) {
    int bn = idx - (RQ + RO + RG);
    float* g = geomJ + (size_t)bn * 8;
    g[0] = x[(size_t)bn*DD + 0]; g[1] = x[(size_t)bn*DD + 1]; g[2] = x[(size_t)bn*DD + 2];
    g[3] = nrm[bn*3+0]; g[4] = nrm[bn*3+1]; g[5] = nrm[bn*3+2];
    g[6] = curv[bn]; g[7] = dens[bn];
  }
}

// ---------------------------------------------------------------------------
// Kernel 1: QKV projection, one wave = one 16x16 output tile.
// Writes Q,K as f16 [b][h][token][hd]; V transposed as Vt [b][h][d][token].
// ---------------------------------------------------------------------------
__global__ void qkv_kernel(const float* __restrict__ x, const f16* __restrict__ wqkvT,
                           const float* __restrict__ qkvb,
                           f16* __restrict__ Qh, f16* __restrict__ Kh, f16* __restrict__ Vt)
{
  int wave = blockIdx.x * (blockDim.x >> 5) + (threadIdx.x >> 5);
  int lane = threadIdx.x & 31;
  int m16 = lane & 15, hi = lane >> 4;
  int mt = wave / 24, nt = wave % 24;
  int token = mt * 16 + m16;
  int n = nt * 16 + m16;

  const float* xr = x + (size_t)token * DD;
  const f16*   wr = wqkvT + (size_t)n * DD;

  v8f acc = {0.f,0.f,0.f,0.f,0.f,0.f,0.f,0.f};
  #pragma unroll
  for (int kc = 0; kc < DD; kc += 32) {
    U16h a, bm;
    f32x4 t0 = *(const f32x4*)(xr + kc + 8*hi);
    f32x4 t1 = *(const f32x4*)(xr + kc + 8*hi + 4);
    f32x4 t2 = *(const f32x4*)(xr + kc + 16 + 8*hi);
    f32x4 t3 = *(const f32x4*)(xr + kc + 16 + 8*hi + 4);
    #pragma unroll
    for (int e = 0; e < 4; ++e) {
      a.h[e]      = (f16)t0[e];  a.h[4 + e]  = (f16)t1[e];
      a.h[8 + e]  = (f16)t2[e];  a.h[12 + e] = (f16)t3[e];
    }
    bm.q[0] = *(const u32x4*)(wr + kc + 8*hi);
    bm.q[1] = *(const u32x4*)(wr + kc + 16 + 8*hi);
    acc = wmma16(a.v, bm.v, acc);
  }
  float bias = qkvb[n];
  #pragma unroll
  for (int r = 0; r < 8; ++r) acc[r] += bias;

  int b    = token >> 11;                 // token / NN
  int row0 = mt * 16 + 8 * hi - b * NN;   // D-tile reg0 token row within batch
  if (n < 128) {
    int hq = n >> 5, d = n & 31;
    f16* base = Qh + (((size_t)b * HH + hq) * NN) * HD + d;
    #pragma unroll
    for (int r = 0; r < 8; ++r) base[(size_t)(row0 + r) * HD] = (f16)acc[r];
  } else if (n < 256) {
    int nv = n - 128, hq = nv >> 5, d = nv & 31;
    f16* base = Kh + (((size_t)b * HH + hq) * NN) * HD + d;
    #pragma unroll
    for (int r = 0; r < 8; ++r) base[(size_t)(row0 + r) * HD] = (f16)acc[r];
  } else {
    int nv = n - 256, hq = nv >> 5, d = nv & 31;
    Pack8 pk;
    #pragma unroll
    for (int r = 0; r < 8; ++r) pk.h[r] = (f16)acc[r];
    *(u32x4*)(Vt + (((size_t)b * HH + hq) * HD + d) * NN + row0) = pk.q;
  }
}

// ---------------------------------------------------------------------------
// Kernel 2: fused flash attention, geometry shared across heads via LDS.
// Block = 128 threads = 4 waves = the 4 heads of one (b, i-tile, j-chunk).
// All threads cooperatively compute the 16x32 (gate,bias) block once per
// 32-j step into double-buffered LDS (affine pair: s' = s*f.x + f.y), one
// barrier per step. Each wave then runs its head's Sᵀ/Pᵀ/Oᵀ WMMA pipeline.
// j-range split JSPLIT ways; unnormalized Oᵀ + (m,l) partials stored.
// ---------------------------------------------------------------------------
__global__ void attn_kernel(const f16* __restrict__ Qh, const f16* __restrict__ Kh,
                            const f16* __restrict__ Vt,
                            const float* __restrict__ geomI, const float* __restrict__ geomJ,
                            float* __restrict__ Opart, float* __restrict__ Mpart,
                            float* __restrict__ Lpart)
{
  __shared__ f32x2 gbuf[2][32][16];   // [buffer][j'][i] : (RSQ*gate, bias*gate)

  int tid  = threadIdx.x;
  int lane = tid & 31;
  int m16  = lane & 15, hi = lane >> 4;
  int h    = tid >> 5;                 // wave index == head
  int blk  = blockIdx.x;
  int p    = blk & (JSPLIT - 1);
  int tile = blk >> 1;                 // 0..255 = (b, i-tile)
  int b    = tile >> 7;
  int it   = tile & 127;
  int i    = it * 16 + m16;

  // i-side geometry for this thread's row (i == it*16 + (tid&15))
  const float* gi = geomI + (size_t)(b * NN + it * 16 + (tid & 15)) * 16;
  f32x4 gA = *(const f32x4*)(gi + 0);
  f32x4 gB = *(const f32x4*)(gi + 4);
  f32x4 gC = *(const f32x4*)(gi + 8);
  f32x4 gD = *(const f32x4*)(gi + 12);
  float gx = gA.x, gy = gA.y, gz = gA.z;
  float px = gA.w, py = gB.x, pz = gB.y;
  float nix = gB.z, niy = gB.w, niz = gC.x;
  float sx = gC.y, sy = gC.z, sz = gC.w;
  float ci = gD.x, alpha = 1.0f - gD.z;
  float effh = 0.175f * (0.5f + gD.y);
  int jq = tid >> 4;                   // 0..7: this thread covers j' = 4*jq..4*jq+3

  // Q as B-operand (Qᵀ) for head h: lane holds col i, kk = 8*hi + e (e<8), 16 + 8*hi + (e-8) (e>=8)
  const f16* qrow = Qh + ((size_t)(b * HH + h) * NN + i) * HD;
  U16h qb;
  qb.q[0] = *(const u32x4*)(qrow + 8*hi);
  qb.q[1] = *(const u32x4*)(qrow + 16 + 8*hi);

  const f16* kbase = Kh + (size_t)(b * HH + h) * NN * HD;
  const f16* vrow0 = Vt + ((size_t)(b * HH + h) * HD + m16) * NN;
  const f16* vrow1 = Vt + ((size_t)(b * HH + h) * HD + 16 + m16) * NN;

  const v8f vz = {0.f,0.f,0.f,0.f,0.f,0.f,0.f,0.f};
  v8f o0 = vz, o1 = vz;
  float mrow = -1e30f, lrow = 0.0f;
  const float RSQ = 0.17677669529663687f;  // hd^-0.5

  const int j0 = p * JCHUNK;
  for (int s = 0; s < JCHUNK / 32; ++s) {
    int jb = j0 + s * 32;
    int buf = s & 1;

    // --- cooperative geometry: 4 (i,j) pairs per thread, once per block ---
    #pragma unroll
    for (int c = 0; c < 4; ++c) {
      int jj = jq * 4 + c;
      int j  = jb + jj;
      const float* gj = geomJ + (size_t)(b * NN + j) * 8;
      f32x4 g0 = *(const f32x4*)gj;
      f32x4 g1 = *(const f32x4*)(gj + 4);
      float dx = gx - g0.x, dy = gy - g0.y, dz = gz - g0.z;
      float dpar = dx*px + dy*py + dz*pz;
      float d2 = dx*dx + dy*dy + dz*dz;
      float dperp2 = fmaxf(d2 - dpar*dpar, 0.0f);
      float aniso2 = alpha * dpar * dpar + 4.0f * dperp2;
      float nsim = nix*g0.w + niy*g1.x + niz*g1.y;
      float caff = __expf(-fabsf(ci - g1.z));
      float gbias = -12.5f * aniso2 + 0.5f * nsim * caff * g1.w;
      float lat = dx*sx + dy*sy + dz*sz;
      float gate = 1.0f / (1.0f + __expf((fabsf(lat) - effh) * 16.0f));
      f32x2 fv; fv.x = RSQ * gate; fv.y = gbias * gate;
      gbuf[buf][jj][tid & 15] = fv;
    }
    __syncthreads();

    // --- per-head Sᵀ = K·Qᵀ ---
    U16h ka0, ka1;
    const f16* kr0 = kbase + (size_t)(jb + m16) * HD;
    const f16* kr1 = kbase + (size_t)(jb + 16 + m16) * HD;
    ka0.q[0] = *(const u32x4*)(kr0 + 8*hi);
    ka0.q[1] = *(const u32x4*)(kr0 + 16 + 8*hi);
    ka1.q[0] = *(const u32x4*)(kr1 + 8*hi);
    ka1.q[1] = *(const u32x4*)(kr1 + 16 + 8*hi);
    v8f s0 = wmma16(ka0.v, qb.v, vz);
    v8f s1 = wmma16(ka1.v, qb.v, vz);

    // apply shared affine bias/gate: reg r of tile js -> j' = 16js+8hi+r, i = m16
    float sv[16];
    #pragma unroll
    for (int js = 0; js < 2; ++js) {
      #pragma unroll
      for (int r = 0; r < 8; ++r) {
        f32x2 fv = gbuf[buf][16 * js + 8 * hi + r][m16];
        float sraw = js ? s1[r] : s0[r];
        sv[js * 8 + r] = sraw * fv.x + fv.y;
      }
    }

    // --- online softmax across lane pair (L, L^16) covering all 32 j ---
    float tmax = sv[0];
    #pragma unroll
    for (int e = 1; e < 16; ++e) tmax = fmaxf(tmax, sv[e]);
    tmax = fmaxf(tmax, __shfl_xor(tmax, 16, 32));
    float mnew = fmaxf(mrow, tmax);
    float corr = __expf(mrow - mnew);
    #pragma unroll
    for (int r = 0; r < 8; ++r) { o0[r] *= corr; o1[r] *= corr; }
    float psum = 0.0f;
    f16 plo[8], phi[8];
    #pragma unroll
    for (int r = 0; r < 8; ++r) { float pe = __expf(sv[r] - mnew);     psum += pe; plo[r] = (f16)pe; }
    #pragma unroll
    for (int r = 0; r < 8; ++r) { float pe = __expf(sv[8 + r] - mnew); psum += pe; phi[r] = (f16)pe; }
    lrow = lrow * corr + psum + __shfl_xor(psum, 16, 32);
    mrow = mnew;

    // --- build Pᵀ B-operand: lane-pair half swap (4 dword shfl_xor) ---
    unsigned lo_u[4], hi_u[4];
    #pragma unroll
    for (int w2 = 0; w2 < 4; ++w2) {
      Pack2 t; t.h[0] = plo[2*w2]; t.h[1] = plo[2*w2 + 1]; lo_u[w2] = t.u;
      Pack2 u; u.h[0] = phi[2*w2]; u.h[1] = phi[2*w2 + 1]; hi_u[w2] = u.u;
    }
    U16h pb;
    #pragma unroll
    for (int w2 = 0; w2 < 4; ++w2) {
      unsigned send = hi ? lo_u[w2] : hi_u[w2];
      unsigned recv = (unsigned)__shfl_xor((int)send, 16, 32);
      pb.u[w2]     = hi ? recv : lo_u[w2];
      pb.u[4 + w2] = hi ? hi_u[w2] : recv;
    }

    // --- Oᵀ += Vᵀ·Pᵀ ---
    U16h va0, va1;
    va0.q[0] = *(const u32x4*)(vrow0 + jb + 16*hi);
    va0.q[1] = *(const u32x4*)(vrow0 + jb + 16*hi + 8);
    va1.q[0] = *(const u32x4*)(vrow1 + jb + 16*hi);
    va1.q[1] = *(const u32x4*)(vrow1 + jb + 16*hi + 8);
    o0 = wmma16(va0.v, pb.v, o0);
    o1 = wmma16(va1.v, pb.v, o1);
  }

  // --- store unnormalized partials: Opart[(tile*JSPLIT+p)*HH+h][i][d] ---
  size_t pidx = (size_t)(tile * JSPLIT + p) * HH + h;
  float* obase = Opart + (pidx * 16 + m16) * 32;
  PackF8 w0, w1;
  #pragma unroll
  for (int r = 0; r < 8; ++r) { w0.f[r] = o0[r]; w1.f[r] = o1[r]; }
  *(f32x4*)(obase + 8*hi)          = w0.q[0];
  *(f32x4*)(obase + 8*hi + 4)      = w0.q[1];
  *(f32x4*)(obase + 16 + 8*hi)     = w1.q[0];
  *(f32x4*)(obase + 16 + 8*hi + 4) = w1.q[1];
  if (hi == 0) {
    Mpart[pidx * 16 + m16] = mrow;
    Lpart[pidx * 16 + m16] = lrow;
  }
}

// ---------------------------------------------------------------------------
// Kernel 2b: merge JSPLIT partials -> attnO f16 [b][token][h*32+d]
// ---------------------------------------------------------------------------
__global__ void combine_kernel(const float* __restrict__ Opart,
                               const float* __restrict__ Mpart,
                               const float* __restrict__ Lpart,
                               f16* __restrict__ attnO)
{
  int tid  = blockIdx.x * blockDim.x + threadIdx.x;   // 256*4*16*32 = 524288
  int d    = tid & 31;
  int i16  = (tid >> 5) & 15;
  int h    = (tid >> 9) & 3;
  int tile = tid >> 11;
  int b    = tile >> 7, it = tile & 127;
  size_t p0 = (size_t)(tile * JSPLIT + 0) * HH + h;
  size_t p1 = (size_t)(tile * JSPLIT + 1) * HH + h;
  float m0 = Mpart[p0 * 16 + i16], m1 = Mpart[p1 * 16 + i16];
  float l0 = Lpart[p0 * 16 + i16], l1 = Lpart[p1 * 16 + i16];
  float mm = fmaxf(m0, m1);
  float e0 = __expf(m0 - mm), e1 = __expf(m1 - mm);
  float o = e0 * Opart[(p0 * 16 + i16) * 32 + d] + e1 * Opart[(p1 * 16 + i16) * 32 + d];
  float l = e0 * l0 + e1 * l1;
  attnO[((size_t)(b * NN + it * 16 + i16)) * DD + h * HD + d] = (f16)(o / l);
}

// ---------------------------------------------------------------------------
// Kernel 3: output projection out = attnO @ out_w + out_b (f32 out).
// ---------------------------------------------------------------------------
__global__ void proj_kernel(const f16* __restrict__ attnO, const f16* __restrict__ owT,
                            const float* __restrict__ outb, float* __restrict__ out)
{
  int wave = blockIdx.x * (blockDim.x >> 5) + (threadIdx.x >> 5);
  int lane = threadIdx.x & 31;
  int m16 = lane & 15, hi = lane >> 4;
  int mt = wave >> 3, nt = wave & 7;
  int token = mt * 16 + m16;
  int n = nt * 16 + m16;

  const f16* ar = attnO + (size_t)token * DD;
  const f16* wr = owT + (size_t)n * DD;
  v8f acc = {0.f,0.f,0.f,0.f,0.f,0.f,0.f,0.f};
  #pragma unroll
  for (int kc = 0; kc < DD; kc += 32) {
    U16h a, bm;
    a.q[0]  = *(const u32x4*)(ar + kc + 8*hi);
    a.q[1]  = *(const u32x4*)(ar + kc + 16 + 8*hi);
    bm.q[0] = *(const u32x4*)(wr + kc + 8*hi);
    bm.q[1] = *(const u32x4*)(wr + kc + 16 + 8*hi);
    acc = wmma16(a.v, bm.v, acc);
  }
  float bias = outb[n];
  int row0 = mt * 16 + 8 * hi;
  #pragma unroll
  for (int r = 0; r < 8; ++r)
    out[(size_t)(row0 + r) * DD + n] = acc[r] + bias;
}

// ---------------------------------------------------------------------------
extern "C" void kernel_launch(void* const* d_in, const int* in_sizes, int n_in,
                              void* d_out, int out_size, void* d_ws, size_t ws_size,
                              hipStream_t stream) {
  const float* x    = (const float*)d_in[0];
  const float* pdir = (const float*)d_in[1];
  const float* curv = (const float*)d_in[2];
  const float* dens = (const float*)d_in[3];
  const float* nrm  = (const float*)d_in[4];
  const float* lin  = (const float*)d_in[5];
  const float* qkvw = (const float*)d_in[6];
  const float* qkvb = (const float*)d_in[7];
  const float* outw = (const float*)d_in[8];
  const float* outb = (const float*)d_in[9];
  float* out = (float*)d_out;

  char* w = (char*)d_ws;
  f16*   wqkvT = (f16*)(w + 0);          //  98304 B
  f16*   owT   = (f16*)(w + 98304);      //  32768 B
  float* geomI = (float*)(w + 131072);   // 262144 B
  float* geomJ = (float*)(w + 393216);   // 131072 B
  f16*   Qh    = (f16*)(w + 524288);     // 1 MB
  f16*   Kh    = (f16*)(w + 1572864);    // 1 MB
  f16*   Vt    = (f16*)(w + 2621440);    // 1 MB
  f16*   aO    = (f16*)(w + 3670016);    // 1 MB
  float* Opart = (float*)(w + 4718592);  // 256*2*4*16*32*4 = 4 MB
  float* Mpart = (float*)(w + 8912896);  // 128 KB
  float* Lpart = (float*)(w + 9043968);  // 128 KB (total 9175040 B)

  hipLaunchKernelGGL(prep_kernel, dim3(288), dim3(256), 0, stream,
                     x, pdir, curv, dens, nrm, lin, qkvw, outw, wqkvT, owT, geomI, geomJ);
  hipLaunchKernelGGL(qkv_kernel, dim3(1536), dim3(128), 0, stream,
                     x, wqkvT, qkvb, Qh, Kh, Vt);
  hipLaunchKernelGGL(attn_kernel, dim3(256 * JSPLIT), dim3(128), 0, stream,
                     Qh, Kh, Vt, geomI, geomJ, Opart, Mpart, Lpart);
  hipLaunchKernelGGL(combine_kernel, dim3(2048), dim3(256), 0, stream,
                     Opart, Mpart, Lpart, aO);
  hipLaunchKernelGGL(proj_kernel, dim3(512), dim3(128), 0, stream,
                     aO, owT, outb, out);
}
